// E_GCL_vel_mechanics_19121194401947
// MI455X (gfx1250) — matrix operand order repaired and checked by reading, not compiled, not run on hardware
//
#include <hip/hip_runtime.h>
#include <hip/hip_bf16.h>
#include <math.h>

typedef _Float16 v16h __attribute__((ext_vector_type(16)));
typedef float    v8f  __attribute__((ext_vector_type(8)));

#define KEF 160   // 129 padded to 5 K-chunks of 32

// ---------------- WMMA fragment loaders (CDNA5 wave32 layouts) ----------------
// A 16x32 f16: lanes 0-15 hold M=lane, K 0..15; lanes 16-31 K 16..31
__device__ __forceinline__ v16h load_a_frag(const _Float16* t, int stride, int kbase, int lane) {
  int m = lane & 15;
  int koff = (lane >> 4) << 4;           // 0 or 16
  const _Float16* p = t + m * stride + kbase + koff;
  v16h a;
#pragma unroll
  for (int i = 0; i < 16; ++i) a[i] = p[i];
  return a;
}

// B 32x16 f16 from a transposed weight tile WT[n][k] (row stride = stride):
// lanes 0-15: N=lane, K=0..15 ; lanes 16-31: N=lane-16, K=16..31
__device__ __forceinline__ v16h load_b_frag(const _Float16* wt, int stride, int kbase, int ntile, int lane) {
  int n = lane & 15;
  int k0 = kbase + ((lane >> 4) << 4);   // +0 or +16
  const _Float16* p = wt + (ntile * 16 + n) * stride + k0;
  v16h b;
#pragma unroll
  for (int i = 0; i < 16; ++i) b[i] = p[i];
  return b;
}

// =============================================================================
// Edge kernel: fused gather -> edge MLP -> coord MLP -> atomic segment sums
// =============================================================================
__global__ __launch_bounds__(256) void edge_kernel(
    const float* __restrict__ h, const int* __restrict__ eidx,
    const float* __restrict__ x,
    const float* __restrict__ ew1, const float* __restrict__ eb1,
    const float* __restrict__ ew2, const float* __restrict__ eb2,
    const float* __restrict__ cw1, const float* __restrict__ cb1,
    const float* __restrict__ cw2,
    float* __restrict__ f_sum, float* __restrict__ cnt, float* __restrict__ agg,
    int E, int numBatches) {
  __shared__ _Float16 sW1T[64 * KEF];     // edge_w1^T  [n][k]
  __shared__ _Float16 sW2T[64 * 64];      // edge_w2^T
  __shared__ _Float16 sCW1T[64 * 64];     // coord_w1^T
  __shared__ float    sCW2[64];
  __shared__ float    sB1[64], sB2[64], sCB1[64];
  __shared__ _Float16 sA[8][16 * KEF];    // per-wave ef_in tile
  __shared__ _Float16 sE1[8][16 * 64];    // layer1 out
  __shared__ _Float16 sEF[8][16 * 64];    // edge_feat
  __shared__ float    sCD[8][16 * 3];
  __shared__ int      sRow[8][16];
  __shared__ float    sCM[8][16];

  const int tid = threadIdx.x;
  for (int i = tid; i < 64 * KEF; i += 256) {
    int n = i / KEF, k = i - n * KEF;
    sW1T[i] = (_Float16)((k < 129) ? ew1[k * 64 + n] : 0.f);
  }
  for (int i = tid; i < 64 * 64; i += 256) {
    int n = i >> 6, k = i & 63;
    sW2T[i]  = (_Float16)ew2[k * 64 + n];
    sCW1T[i] = (_Float16)cw1[k * 64 + n];
  }
  if (tid < 64) { sCW2[tid] = cw2[tid]; sB1[tid] = eb1[tid]; sB2[tid] = eb2[tid]; sCB1[tid] = cb1[tid]; }
  __syncthreads();

  const int lane = tid & 31;
  const int w    = tid >> 5;
  const int e16  = lane & 15;
  const int part = lane >> 4;
  const int nn   = lane & 15;
  const int half = lane >> 4;

  const int* __restrict__ rowp = eidx;
  const int* __restrict__ colp = eidx + E;

  for (int batch = blockIdx.x; batch < numBatches; batch += gridDim.x) {
    const int base = batch * 128 + w * 16;

    // ---------------- gather ef_in = [h[row] | h[col] | radial | 0-pad] ------
    {
      int eI = base + e16;
      int ec = (eI < E) ? eI : (E - 1);
      int r = rowp[ec], c = colp[ec];
      const float* hr = h + (size_t)r * 64 + part * 32;
      const float* hc = h + (size_t)c * 64 + part * 32;
      _Float16* arow = &sA[w][e16 * KEF];
#pragma unroll
      for (int k = 0; k < 32; ++k) arow[part * 32 + k]      = (_Float16)hr[k];
#pragma unroll
      for (int k = 0; k < 32; ++k) arow[64 + part * 32 + k] = (_Float16)hc[k];
      if (part == 0) {
        float d0 = x[(size_t)r*3+0] - x[(size_t)c*3+0];
        float d1 = x[(size_t)r*3+1] - x[(size_t)c*3+1];
        float d2 = x[(size_t)r*3+2] - x[(size_t)c*3+2];
        arow[128] = (_Float16)(d0*d0 + d1*d1 + d2*d2);
#pragma unroll
        for (int k = 129; k < KEF; ++k) arow[k] = (_Float16)0.f;
        sCD[w][e16*3+0] = d0; sCD[w][e16*3+1] = d1; sCD[w][e16*3+2] = d2;
        sRow[w][e16] = r;
      }
    }
    __syncthreads();

    // ---------------- layer 1: ef_in(16x160) @ W1(160x64) + b1, relu ---------
    {
      v8f acc[4];
#pragma unroll
      for (int nt = 0; nt < 4; ++nt) {
        float bv = sB1[nt * 16 + nn];
#pragma unroll
        for (int i = 0; i < 8; ++i) acc[nt][i] = bv;
      }
#pragma unroll
      for (int kc = 0; kc < 5; ++kc) {
        v16h a = load_a_frag(&sA[w][0], KEF, kc * 32, lane);
#pragma unroll
        for (int nt = 0; nt < 4; ++nt) {
          v16h b = load_b_frag(sW1T, KEF, kc * 32, nt, lane);
          acc[nt] = __builtin_amdgcn_wmma_f32_16x16x32_f16(false, a, false, b,
                                                           (short)0, acc[nt], false, false);
        }
      }
#pragma unroll
      for (int nt = 0; nt < 4; ++nt)
#pragma unroll
        for (int i = 0; i < 8; ++i) {
          int m = half * 8 + i;
          sE1[w][m * 64 + nt * 16 + nn] = (_Float16)fmaxf(acc[nt][i], 0.f);
        }
    }
    __syncthreads();

    // ---------------- layer 2: e1 @ W2 + b2, relu -> edge_feat ---------------
    {
      v8f acc[4];
#pragma unroll
      for (int nt = 0; nt < 4; ++nt) {
        float bv = sB2[nt * 16 + nn];
#pragma unroll
        for (int i = 0; i < 8; ++i) acc[nt][i] = bv;
      }
#pragma unroll
      for (int kc = 0; kc < 2; ++kc) {
        v16h a = load_a_frag(&sE1[w][0], 64, kc * 32, lane);
#pragma unroll
        for (int nt = 0; nt < 4; ++nt) {
          v16h b = load_b_frag(sW2T, 64, kc * 32, nt, lane);
          acc[nt] = __builtin_amdgcn_wmma_f32_16x16x32_f16(false, a, false, b,
                                                           (short)0, acc[nt], false, false);
        }
      }
#pragma unroll
      for (int nt = 0; nt < 4; ++nt)
#pragma unroll
        for (int i = 0; i < 8; ++i) {
          int m = half * 8 + i;
          float vv = fmaxf(acc[nt][i], 0.f);
          sEF[w][m * 64 + nt * 16 + nn] = (_Float16)vv;
          int eI = base + m;
          if (eI < E) {
            int r = sRow[w][m];
            unsafeAtomicAdd(&agg[(size_t)r * 64 + nt * 16 + nn], vv);
          }
        }
    }
    __syncthreads();

    // ---------------- coord MLP: relu(ef @ CW1 + cb1) @ cw2 -> cm ------------
    {
      v8f acc[4];
#pragma unroll
      for (int nt = 0; nt < 4; ++nt) {
        float bv = sCB1[nt * 16 + nn];
#pragma unroll
        for (int i = 0; i < 8; ++i) acc[nt][i] = bv;
      }
#pragma unroll
      for (int kc = 0; kc < 2; ++kc) {
        v16h a = load_a_frag(&sEF[w][0], 64, kc * 32, lane);
#pragma unroll
        for (int nt = 0; nt < 4; ++nt) {
          v16h b = load_b_frag(sCW1T, 64, kc * 32, nt, lane);
          acc[nt] = __builtin_amdgcn_wmma_f32_16x16x32_f16(false, a, false, b,
                                                           (short)0, acc[nt], false, false);
        }
      }
      float pm[8];
#pragma unroll
      for (int i = 0; i < 8; ++i) pm[i] = 0.f;
#pragma unroll
      for (int nt = 0; nt < 4; ++nt) {
        float wv = sCW2[nt * 16 + nn];
#pragma unroll
        for (int i = 0; i < 8; ++i) pm[i] = fmaf(fmaxf(acc[nt][i], 0.f), wv, pm[i]);
      }
#pragma unroll
      for (int i = 0; i < 8; ++i) {
        float s = pm[i];
#pragma unroll
        for (int msk = 1; msk < 16; msk <<= 1) s += __shfl_xor(s, msk, 32);
        pm[i] = s;
      }
      if (nn == 0) {
#pragma unroll
        for (int i = 0; i < 8; ++i) sCM[w][half * 8 + i] = pm[i];
      }
    }
    __syncthreads();

    // ---------------- trans + f_sum / cnt atomics ----------------------------
    if (lane < 16) {
      int eI = base + lane;
      if (eI < E) {
        int r = sRow[w][lane];
        float cmv = sCM[w][lane];
        unsafeAtomicAdd(&cnt[r], 1.f);
#pragma unroll
        for (int j = 0; j < 3; ++j) {
          float t = sCD[w][lane * 3 + j] * cmv;
          t = fminf(fmaxf(t, -100.f), 100.f);
          unsafeAtomicAdd(&f_sum[(size_t)r * 3 + j], t);
        }
      }
    }
    __syncthreads();
  }
}

// =============================================================================
// f = f_sum / max(cnt, 1)
// =============================================================================
__global__ void finalize_f(const float* __restrict__ f_sum, const float* __restrict__ cnt,
                           float* __restrict__ f, int N) {
  int i = blockIdx.x * blockDim.x + threadIdx.x;
  if (i < N) {
    float c = 1.f / fmaxf(cnt[i], 1.f);
    f[(size_t)i*3+0] = f_sum[(size_t)i*3+0] * c;
    f[(size_t)i*3+1] = f_sum[(size_t)i*3+1] * c;
    f[(size_t)i*3+2] = f_sum[(size_t)i*3+2] * c;
  }
}

// =============================================================================
// Node MLP: nm=[agg/cnt | h | agg] (192) -> 64 relu -> 64 ; h_out = h + out
// =============================================================================
__global__ __launch_bounds__(256) void node_kernel(
    const float* __restrict__ h, const float* __restrict__ agg, const float* __restrict__ cnt,
    const float* __restrict__ nw1, const float* __restrict__ nb1,
    const float* __restrict__ nw2, const float* __restrict__ nb2,
    float* __restrict__ hout, int N) {
  __shared__ _Float16 sW1T[64 * 192];
  __shared__ _Float16 sW2T[64 * 64];
  __shared__ float    sB1[64], sB2[64];
  __shared__ _Float16 sA[8][16 * 192];
  __shared__ _Float16 sE1[8][16 * 64];

  const int tid = threadIdx.x;
  for (int i = tid; i < 64 * 192; i += 256) {
    int n = i / 192, k = i - n * 192;
    sW1T[i] = (_Float16)nw1[k * 64 + n];
  }
  for (int i = tid; i < 64 * 64; i += 256) {
    int n = i >> 6, k = i & 63;
    sW2T[i] = (_Float16)nw2[k * 64 + n];
  }
  if (tid < 64) { sB1[tid] = nb1[tid]; sB2[tid] = nb2[tid]; }
  __syncthreads();

  const int lane = tid & 31;
  const int w    = tid >> 5;
  const int e16  = lane & 15;
  const int part = lane >> 4;
  const int nn   = lane & 15;
  const int half = lane >> 4;
  const int base = blockIdx.x * 128 + w * 16;

  // gather nm tile
  {
    int nI = base + e16;
    int nc = (nI < N) ? nI : (N - 1);
    float cinv = 1.f / fmaxf(cnt[nc], 1.f);
    const float* ag = agg + (size_t)nc * 64 + part * 32;
    const float* hp = h   + (size_t)nc * 64 + part * 32;
    _Float16* arow = &sA[w][e16 * 192];
#pragma unroll
    for (int k = 0; k < 32; ++k) {
      float a = ag[k];
      arow[part * 32 + k]        = (_Float16)(a * cinv);   // others
      arow[64 + part * 32 + k]   = (_Float16)hp[k];        // h
      arow[128 + part * 32 + k]  = (_Float16)a;            // agg
    }
  }
  __syncthreads();

  // layer 1
  {
    v8f acc[4];
#pragma unroll
    for (int nt = 0; nt < 4; ++nt) {
      float bv = sB1[nt * 16 + nn];
#pragma unroll
      for (int i = 0; i < 8; ++i) acc[nt][i] = bv;
    }
#pragma unroll
    for (int kc = 0; kc < 6; ++kc) {
      v16h a = load_a_frag(&sA[w][0], 192, kc * 32, lane);
#pragma unroll
      for (int nt = 0; nt < 4; ++nt) {
        v16h b = load_b_frag(sW1T, 192, kc * 32, nt, lane);
        acc[nt] = __builtin_amdgcn_wmma_f32_16x16x32_f16(false, a, false, b,
                                                         (short)0, acc[nt], false, false);
      }
    }
#pragma unroll
    for (int nt = 0; nt < 4; ++nt)
#pragma unroll
      for (int i = 0; i < 8; ++i) {
        int m = half * 8 + i;
        sE1[w][m * 64 + nt * 16 + nn] = (_Float16)fmaxf(acc[nt][i], 0.f);
      }
  }
  __syncthreads();

  // layer 2 + residual
  {
    v8f acc[4];
#pragma unroll
    for (int nt = 0; nt < 4; ++nt) {
      float bv = sB2[nt * 16 + nn];
#pragma unroll
      for (int i = 0; i < 8; ++i) acc[nt][i] = bv;
    }
#pragma unroll
    for (int kc = 0; kc < 2; ++kc) {
      v16h a = load_a_frag(&sE1[w][0], 64, kc * 32, lane);
#pragma unroll
      for (int nt = 0; nt < 4; ++nt) {
        v16h b = load_b_frag(sW2T, 64, kc * 32, nt, lane);
        acc[nt] = __builtin_amdgcn_wmma_f32_16x16x32_f16(false, a, false, b,
                                                         (short)0, acc[nt], false, false);
      }
    }
#pragma unroll
    for (int nt = 0; nt < 4; ++nt)
#pragma unroll
      for (int i = 0; i < 8; ++i) {
        int m = half * 8 + i;
        int node = base + m;
        if (node < N) {
          int col = nt * 16 + nn;
          hout[(size_t)node * 64 + col] = acc[nt][i] + h[(size_t)node * 64 + col];
        }
      }
  }
}

// =============================================================================
// Scalar mechanics helpers (f32; negligible FLOPs)
// =============================================================================
__device__ __forceinline__ void cross3(const float* a, const float* b, float* o) {
  o[0] = a[1]*b[2] - a[2]*b[1];
  o[1] = a[2]*b[0] - a[0]*b[2];
  o[2] = a[0]*b[1] - a[1]*b[0];
}

__device__ __forceinline__ void rot_apply(const float* wv, float* r) {
  float n2 = wv[0]*wv[0] + wv[1]*wv[1] + wv[2]*wv[2];
  float nw = sqrtf(n2);
  float inv = 1.f / fmaxf(nw, 1e-12f);
  float dx = wv[0]*inv, dy = wv[1]*inv, dz = wv[2]*inv;
  float c = cosf(nw), s = sinf(nw), C = 1.f - c;
  float r0 = r[0], r1 = r[1], r2 = r[2];
  float o0 = (c + C*dx*dx)*r0 + (C*dx*dy - s*dz)*r1 + (C*dx*dz + s*dy)*r2;
  float o1 = (C*dx*dy + s*dz)*r0 + (c + C*dy*dy)*r1 + (C*dy*dz - s*dx)*r2;
  float o2 = (C*dx*dz - s*dy)*r0 + (C*dy*dz + s*dx)*r1 + (c + C*dz*dz)*r2;
  r[0] = o0; r[1] = o1; r[2] = o2;
}

// 64 -> 64(relu) -> 1 MLP on a register-resident input; weights via uniform (scalar) loads
__device__ __forceinline__ float mlp_64_1(const float* __restrict__ in,
    const float* __restrict__ w1, const float* __restrict__ b1,
    const float* __restrict__ w2, const float* __restrict__ b2) {
  float out = b2[0];
  for (int j = 0; j < 64; ++j) {
    float a = b1[j];
#pragma unroll
    for (int k = 0; k < 64; ++k) a = fmaf(in[k], w1[k*64 + j], a);
    out = fmaf(fmaxf(a, 0.f), w2[j], out);
  }
  return out;
}

// out[64] = center_mlp(h1) + center_mlp(h2)
__device__ __forceinline__ void center2(const float* __restrict__ h1, const float* __restrict__ h2,
    const float* __restrict__ w1, const float* __restrict__ b1,
    const float* __restrict__ w2, const float* __restrict__ b2, float* __restrict__ out) {
#pragma unroll
  for (int d = 0; d < 64; ++d) out[d] = 2.f * b2[d];
  for (int j = 0; j < 64; ++j) {
    float a1 = b1[j], a2 = b1[j];
#pragma unroll
    for (int k = 0; k < 64; ++k) {
      float wv = w1[k*64 + j];
      a1 = fmaf(h1[k], wv, a1);
      a2 = fmaf(h2[k], wv, a2);
    }
    float hs = fmaxf(a1, 0.f) + fmaxf(a2, 0.f);
#pragma unroll
    for (int d = 0; d < 64; ++d) out[d] = fmaf(hs, w2[j*64 + d], out[d]);
  }
}

// =============================================================================
// iso: vi = vel_mlp(h)*v + f ; x += vi
// =============================================================================
__global__ void iso_kernel(const float* __restrict__ h, const float* __restrict__ x,
                           const float* __restrict__ v, const float* __restrict__ f,
                           const int* __restrict__ iso, int nIso,
                           const float* vw1, const float* vb1, const float* vw2, const float* vb2,
                           float* __restrict__ xo, float* __restrict__ vo) {
  int t = blockIdx.x * blockDim.x + threadIdx.x;
  if (t >= nIso) return;
  int n = iso[t];
  float hr[64];
#pragma unroll
  for (int k = 0; k < 64; ++k) hr[k] = h[(size_t)n*64 + k];
  float s = mlp_64_1(hr, vw1, vb1, vw2, vb2);
#pragma unroll
  for (int j = 0; j < 3; ++j) {
    float vi = s * v[(size_t)n*3 + j] + f[(size_t)n*3 + j];
    xo[(size_t)n*3 + j] = x[(size_t)n*3 + j] + vi;
    vo[(size_t)n*3 + j] = vi;
  }
}

// =============================================================================
// stick rigid-pair mechanics
// =============================================================================
__global__ void stick_kernel(const float* __restrict__ h, const float* __restrict__ x,
                             const float* __restrict__ v, const float* __restrict__ f,
                             const int* __restrict__ sidx, int nS,
                             const float* vw1, const float* vb1, const float* vw2, const float* vb2,
                             const float* ww1, const float* wb1, const float* ww2, const float* wb2,
                             const float* cw1, const float* cb1, const float* cw2m, const float* cb2,
                             const float* sw1, const float* sb1, const float* sw2, const float* sb2,
                             float* __restrict__ xo, float* __restrict__ vo) {
  int t = blockIdx.x * blockDim.x + threadIdx.x;
  if (t >= nS) return;
  int s1 = sidx[t*2+0], s2 = sidx[t*2+1];
  float x1[3], x2[3], v1[3], v2[3], f1[3], f2[3], x0[3], v0[3];
#pragma unroll
  for (int j = 0; j < 3; ++j) {
    x1[j] = x[(size_t)s1*3+j]; x2[j] = x[(size_t)s2*3+j];
    v1[j] = v[(size_t)s1*3+j]; v2[j] = v[(size_t)s2*3+j];
    f1[j] = f[(size_t)s1*3+j]; f2[j] = f[(size_t)s2*3+j];
    x0[j] = 0.5f*(x1[j]+x2[j]); v0[j] = 0.5f*(v1[j]+v2[j]);
  }
  // stick_f: msg scalar on l2norm(|f|^2)
  auto stickmsg = [&](const float* ff) -> float {
    float inv = ff[0]*ff[0] + ff[1]*ff[1] + ff[2]*ff[2];
    float invn = inv / fmaxf(fabsf(inv), 1e-12f);
    float out = sb2[0];
    for (int j = 0; j < 64; ++j) {
      float a = fmaf(invn, sw1[j], sb1[j]);
      out = fmaf(fmaxf(a, 0.f), sw2[j], out);
    }
    return out;
  };
  float m1 = stickmsg(f1), m2 = stickmsg(f2);
  float a0[3];
#pragma unroll
  for (int j = 0; j < 3; ++j) a0[j] = 0.5f*(f1[j]*m1 + f2[j]*m2);

  float J = 0.f, d1v[3], d2v[3];
#pragma unroll
  for (int j = 0; j < 3; ++j) {
    d1v[j] = x1[j]-x0[j]; d2v[j] = x2[j]-x0[j];
    J += d1v[j]*d1v[j] + d2v[j]*d2v[j];
  }
  float c1[3], c2[3], beta[3];
  cross3(d1v, f1, c1); cross3(d2v, f2, c2);
#pragma unroll
  for (int j = 0; j < 3; ++j) beta[j] = (c1[j]+c2[j]) / J;

  float r[3], vr[3];
#pragma unroll
  for (int j = 0; j < 3; ++j) { r[j] = 0.5f*(x1[j]-x2[j]); vr[j] = 0.5f*(v1[j]-v2[j]); }
  float rn = sqrtf(r[0]*r[0]+r[1]*r[1]+r[2]*r[2]);
  float rl[3];
#pragma unroll
  for (int j = 0; j < 3; ++j) rl[j] = r[j] / fmaxf(rn, 1e-12f);
  float wv[3]; cross3(rl, vr, wv);
  float dn = 1.f / fmaxf(rn, 1e-5f);
#pragma unroll
  for (int j = 0; j < 3; ++j) wv[j] *= dn;

  float h1r[64], h2r[64], hc[64];
#pragma unroll
  for (int k = 0; k < 64; ++k) { h1r[k] = h[(size_t)s1*64+k]; h2r[k] = h[(size_t)s2*64+k]; }
  center2(h1r, h2r, cw1, cb1, cw2m, cb2, hc);
  float wm = mlp_64_1(hc, ww1, wb1, ww2, wb2);
  float vm = mlp_64_1(hc, vw1, vb1, vw2, vb2);
#pragma unroll
  for (int j = 0; j < 3; ++j) {
    wv[j] = wm*wv[j] + beta[j];
    v0[j] = vm*v0[j] + a0[j];
    x0[j] += v0[j];
  }
  rot_apply(wv, r);
  float wxr[3]; cross3(wv, r, wxr);
#pragma unroll
  for (int j = 0; j < 3; ++j) {
    xo[(size_t)s1*3+j] = x0[j] + r[j];
    xo[(size_t)s2*3+j] = x0[j] - r[j];
    vo[(size_t)s1*3+j] = v0[j] + wxr[j];
    vo[(size_t)s2*3+j] = v0[j] - wxr[j];
  }
}

// =============================================================================
// hinge 3-body mechanics
// =============================================================================
__global__ void hinge_kernel(const float* __restrict__ h, const float* __restrict__ x,
                             const float* __restrict__ v, const float* __restrict__ f,
                             const int* __restrict__ hidx, int nH,
                             const float* vw1, const float* vb1, const float* vw2, const float* vb2,
                             const float* ww1, const float* wb1, const float* ww2, const float* wb2,
                             const float* cw1, const float* cb1, const float* cw2m, const float* cb2,
                             const float* hw1, const float* hb1, const float* hw2, const float* hb2,
                             float* __restrict__ xo, float* __restrict__ vo) {
  int t = blockIdx.x * blockDim.x + threadIdx.x;
  if (t >= nH) return;
  int g0 = hidx[t*3+0], g1 = hidx[t*3+1], g2 = hidx[t*3+2];
  float hx0[3], hv0[3], hf0[3], hx1[3], hv1[3], hf1[3], hx2[3], hv2[3], hf2[3];
#pragma unroll
  for (int j = 0; j < 3; ++j) {
    hx0[j] = x[(size_t)g0*3+j]; hv0[j] = v[(size_t)g0*3+j]; hf0[j] = f[(size_t)g0*3+j];
    hx1[j] = x[(size_t)g1*3+j]; hv1[j] = v[(size_t)g1*3+j]; hf1[j] = f[(size_t)g1*3+j];
    hx2[j] = x[(size_t)g2*3+j]; hv2[j] = v[(size_t)g2*3+j]; hf2[j] = f[(size_t)g2*3+j];
  }
  auto dot3 = [](const float* a, const float* b) { return a[0]*b[0]+a[1]*b[1]+a[2]*b[2]; };
  auto hingef = [&](const float* cx, const float* cv, const float* cf, float* out3) {
    float c0[3], c1[3], c2[3];
#pragma unroll
    for (int j = 0; j < 3; ++j) { c0[j] = cf[j]; c1[j] = cx[j]-hx0[j]; c2[j] = cv[j]-hv0[j]; }
    float inv[9];
    inv[0]=dot3(c0,c0); inv[1]=dot3(c0,c1); inv[2]=dot3(c0,c2);
    inv[3]=inv[1];      inv[4]=dot3(c1,c1); inv[5]=dot3(c1,c2);
    inv[6]=inv[2];      inv[7]=inv[5];      inv[8]=dot3(c2,c2);
    float n2 = 0.f;
#pragma unroll
    for (int i = 0; i < 9; ++i) n2 += inv[i]*inv[i];
    float sc = 1.f / fmaxf(sqrtf(n2), 1e-12f);
#pragma unroll
    for (int i = 0; i < 9; ++i) inv[i] *= sc;
    float m0 = hb2[0], m1 = hb2[1], m2 = hb2[2];
    for (int j = 0; j < 64; ++j) {
      float a = hb1[j];
#pragma unroll
      for (int i = 0; i < 9; ++i) a = fmaf(inv[i], hw1[i*64 + j], a);
      float hid = fmaxf(a, 0.f);
      m0 = fmaf(hid, hw2[j*3+0], m0);
      m1 = fmaf(hid, hw2[j*3+1], m1);
      m2 = fmaf(hid, hw2[j*3+2], m2);
    }
#pragma unroll
    for (int j = 0; j < 3; ++j) out3[j] = c0[j]*m0 + c1[j]*m1 + c2[j]*m2;
  };
  float t0[3], t1[3], t2[3], a0[3];
  hingef(hx0, hv0, hf0, t0);
  hingef(hx1, hv1, hf1, t1);
  hingef(hx2, hv2, hf2, t2);
#pragma unroll
  for (int j = 0; j < 3; ++j) a0[j] = (t0[j]+t1[j]+t2[j]) * (1.f/3.f);

  auto apply_g = [&](const float* cx, const float* cf, float* out3) {
    float d[3], fm[3];
#pragma unroll
    for (int j = 0; j < 3; ++j) { d[j] = cx[j]-hx0[j]; fm[j] = cf[j]-a0[j]; }
    cross3(d, fm, out3);
    float dn = dot3(d, d);
#pragma unroll
    for (int j = 0; j < 3; ++j) out3[j] /= dn;
  };
  float beta1[3], beta2[3];
  apply_g(hx1, hf1, beta1);
  apply_g(hx2, hf2, beta2);

  auto cmet = [&](const float* cx, const float* cv, float* cr, float* cw) {
    float rv[3];
#pragma unroll
    for (int j = 0; j < 3; ++j) { cr[j] = cx[j]-hx0[j]; rv[j] = cv[j]-hv0[j]; }
    float rn = sqrtf(dot3(cr, cr));
    float rl[3];
#pragma unroll
    for (int j = 0; j < 3; ++j) rl[j] = cr[j] / fmaxf(rn, 1e-12f);
    cross3(rl, rv, cw);
    float dn = 1.f / fmaxf(rn, 1e-5f);
#pragma unroll
    for (int j = 0; j < 3; ++j) cw[j] *= dn;
  };
  float r1[3], w1[3], r2[3], w2[3];
  cmet(hx1, hv1, r1, w1);
  cmet(hx2, hv2, r2, w2);

  float hh1[64], hh2[64], hc[64];
#pragma unroll
  for (int k = 0; k < 64; ++k) { hh1[k] = h[(size_t)g1*64+k]; hh2[k] = h[(size_t)g2*64+k]; }
  center2(hh1, hh2, cw1, cb1, cw2m, cb2, hc);
  float vm = mlp_64_1(hc, vw1, vb1, vw2, vb2);
  float nv0[3], nx0[3];
#pragma unroll
  for (int j = 0; j < 3; ++j) { nv0[j] = vm*hv0[j] + a0[j]; nx0[j] = hx0[j] + nv0[j]; }

  float wm1 = mlp_64_1(hh1, ww1, wb1, ww2, wb2);
  float wm2 = mlp_64_1(hh2, ww1, wb1, ww2, wb2);
#pragma unroll
  for (int j = 0; j < 3; ++j) { w1[j] = wm1*w1[j] + beta1[j]; w2[j] = wm2*w2[j] + beta2[j]; }
  rot_apply(w1, r1);
  rot_apply(w2, r2);
  float wx1[3], wx2[3];
  cross3(w1, r1, wx1);
  cross3(w2, r2, wx2);
#pragma unroll
  for (int j = 0; j < 3; ++j) {
    xo[(size_t)g0*3+j] = nx0[j];
    xo[(size_t)g1*3+j] = nx0[j] + r1[j];
    xo[(size_t)g2*3+j] = nx0[j] + r2[j];
    vo[(size_t)g0*3+j] = nv0[j];
    vo[(size_t)g1*3+j] = nv0[j] + wx1[j];
    vo[(size_t)g2*3+j] = nv0[j] + wx2[j];
  }
}

// =============================================================================
extern "C" void kernel_launch(void* const* d_in, const int* in_sizes, int n_in,
                              void* d_out, int out_size, void* d_ws, size_t ws_size,
                              hipStream_t stream) {
  (void)n_in; (void)out_size; (void)ws_size;
  const float* h     = (const float*)d_in[0];
  const int*   eidx  = (const int*)  d_in[1];
  const float* x     = (const float*)d_in[2];
  const float* v     = (const float*)d_in[3];
  const int*   iso   = (const int*)  d_in[4];
  const int*   stick = (const int*)  d_in[5];
  const int*   hinge = (const int*)  d_in[6];
  const float* ew1 = (const float*)d_in[7];  const float* eb1 = (const float*)d_in[8];
  const float* ew2 = (const float*)d_in[9];  const float* eb2 = (const float*)d_in[10];
  const float* nw1 = (const float*)d_in[11]; const float* nb1 = (const float*)d_in[12];
  const float* nw2 = (const float*)d_in[13]; const float* nb2 = (const float*)d_in[14];
  const float* cw1 = (const float*)d_in[15]; const float* cb1 = (const float*)d_in[16];
  const float* cw2 = (const float*)d_in[17];
  const float* vw1 = (const float*)d_in[18]; const float* vb1 = (const float*)d_in[19];
  const float* vw2 = (const float*)d_in[20]; const float* vb2 = (const float*)d_in[21];
  const float* ww1 = (const float*)d_in[22]; const float* wb1 = (const float*)d_in[23];
  const float* ww2 = (const float*)d_in[24]; const float* wb2 = (const float*)d_in[25];
  const float* crw1 = (const float*)d_in[26]; const float* crb1 = (const float*)d_in[27];
  const float* crw2 = (const float*)d_in[28]; const float* crb2 = (const float*)d_in[29];
  const float* sw1 = (const float*)d_in[30]; const float* sb1 = (const float*)d_in[31];
  const float* sw2 = (const float*)d_in[32]; const float* sb2 = (const float*)d_in[33];
  const float* hw1 = (const float*)d_in[34]; const float* hb1 = (const float*)d_in[35];
  const float* hw2 = (const float*)d_in[36]; const float* hb2 = (const float*)d_in[37];

  const int N = in_sizes[0] / 64;
  const int E = in_sizes[1] / 2;
  const int nIso   = in_sizes[4];
  const int nStick = in_sizes[5] / 2;
  const int nHinge = in_sizes[6] / 3;

  float* f_sum = (float*)d_ws;                 // N*3
  float* cnt   = f_sum + (size_t)N * 3;        // N
  float* agg   = cnt + N;                      // N*64
  float* f     = agg + (size_t)N * 64;         // N*3

  hipMemsetAsync(d_ws, 0, (size_t)N * 68 * sizeof(float), stream);

  float* hout = (float*)d_out;
  float* xo = hout + (size_t)N * 64;
  float* vo = xo + (size_t)N * 3;
  hipMemcpyAsync(xo, (const void*)x, (size_t)N * 3 * sizeof(float), hipMemcpyDeviceToDevice, stream);
  hipMemcpyAsync(vo, (const void*)v, (size_t)N * 3 * sizeof(float), hipMemcpyDeviceToDevice, stream);

  const int numBatches = (E + 127) / 128;
  const int edgeBlocks = numBatches < 1024 ? numBatches : 1024;
  edge_kernel<<<edgeBlocks, 256, 0, stream>>>(h, eidx, x, ew1, eb1, ew2, eb2,
                                              cw1, cb1, cw2, f_sum, cnt, agg, E, numBatches);

  finalize_f<<<(N + 255) / 256, 256, 0, stream>>>(f_sum, cnt, f, N);

  iso_kernel<<<(nIso + 127) / 128, 128, 0, stream>>>(h, x, v, f, iso, nIso,
                                                     vw1, vb1, vw2, vb2, xo, vo);
  stick_kernel<<<(nStick + 63) / 64, 64, 0, stream>>>(h, x, v, f, stick, nStick,
                                                      vw1, vb1, vw2, vb2,
                                                      ww1, wb1, ww2, wb2,
                                                      crw1, crb1, crw2, crb2,
                                                      sw1, sb1, sw2, sb2, xo, vo);
  hinge_kernel<<<(nHinge + 63) / 64, 64, 0, stream>>>(h, x, v, f, hinge, nHinge,
                                                      vw1, vb1, vw2, vb2,
                                                      ww1, wb1, ww2, wb2,
                                                      crw1, crb1, crw2, crb2,
                                                      hw1, hb1, hw2, hb2, xo, vo);

  node_kernel<<<(N + 127) / 128, 256, 0, stream>>>(h, agg, cnt, nw1, nb1, nw2, nb2, hout, N);
}
